// Mamba2GestureRecognizer_47811575939626
// MI455X (gfx1250) — compile-verified, matrix operand order checked by reading, not run on hardware
//
#include <hip/hip_runtime.h>
#include <hip/hip_bf16.h>

#define BATCH      8
#define SEQLEN     2048
#define ROWS       (BATCH*SEQLEN)       /* 16384 */
#define D_MODEL    256
#define INPUT_DIM  63
#define KPAD       64
#define NUM_CLASSES 13
#define NUM_LAYERS 4
#define D_STATE    128
#define D_CONV     4
#define D_INNER    512
#define HEADDIM    64
#define NHEADS     8
#define CONV_DIM   768                   /* D_INNER + 2*D_STATE */
#define D_IN_PROJ  1288                  /* 2*D_INNER + 2*D_STATE + NHEADS */

typedef __bf16 v16bf __attribute__((ext_vector_type(16)));
typedef float  v8f   __attribute__((ext_vector_type(8)));
typedef int    v4i   __attribute__((ext_vector_type(4)));

union FragAB { v16bf bf; v4i chunk[2]; };

__device__ __forceinline__ unsigned short f2bf(float f) {
  unsigned int u = __float_as_uint(f);
  u += 0x7FFFu + ((u >> 16) & 1u);            // round-to-nearest-even
  return (unsigned short)(u >> 16);
}
__device__ __forceinline__ float siluf(float x) { return x / (1.f + __expf(-x)); }

__device__ __forceinline__ float block_sum_256(float v, float* red, int tid) {
  red[tid] = v; __syncthreads();
  #pragma unroll
  for (int s = 128; s > 0; s >>= 1) { if (tid < s) red[tid] += red[tid + s]; __syncthreads(); }
  float r = red[0]; __syncthreads();
  return r;
}

// ---------------------------------------------------------------------------
// C[M,N] f32 = A[M,K]bf16 (row-major) x B[N,K]bf16 (row-major, i.e. B^T).
// Block = 8 waves (wave32). Wave grid 4x2 -> block tile 128(M) x 64(N).
// Each wave: 2x2 WMMA 16x16 tiles, K stepped by 32 (v_wmma_f32_16x16x32_bf16).
// M must be a multiple of 128 (ROWS=16384 ok); N guarded/clamped; K mult of 32.
// ---------------------------------------------------------------------------
__global__ __launch_bounds__(256)
void gemm_bf16_wmma(const unsigned short* __restrict__ A,
                    const unsigned short* __restrict__ Bw,
                    float* __restrict__ C, int N, int K) {
  const int lane = threadIdx.x & 31;
  const int wave = threadIdx.x >> 5;
  const int wm = wave & 3, wn = wave >> 2;
  const int m0 = blockIdx.y * 128 + wm * 32;
  const int n0 = blockIdx.x * 64  + wn * 32;
  const int r  = lane & 15;
  const int hi = lane >> 4;   // 0: K+0..7 / K+16..23 ; 1: K+8..15 / K+24..31

  const v4i* pa0 = (const v4i*)(A + (size_t)(m0 + r)      * K);
  const v4i* pa1 = (const v4i*)(A + (size_t)(m0 + 16 + r) * K);
  int cb0 = n0 + r;      if (cb0 > N - 1) cb0 = N - 1;
  int cb1 = n0 + 16 + r; if (cb1 > N - 1) cb1 = N - 1;
  const v4i* pb0 = (const v4i*)(Bw + (size_t)cb0 * K);
  const v4i* pb1 = (const v4i*)(Bw + (size_t)cb1 * K);

  v8f c00 = {}, c01 = {}, c10 = {}, c11 = {};
  for (int k = 0; k < K; k += 32) {
    const int idx = (k >> 3) + hi;                 // v4i = 8 bf16 elements
    FragAB a0, a1, b0, b1;
    a0.chunk[0] = pa0[idx]; a0.chunk[1] = pa0[idx + 2];
    a1.chunk[0] = pa1[idx]; a1.chunk[1] = pa1[idx + 2];
    b0.chunk[0] = pb0[idx]; b0.chunk[1] = pb0[idx + 2];
    b1.chunk[0] = pb1[idx]; b1.chunk[1] = pb1[idx + 2];
    __builtin_prefetch((const void*)(pa0 + idx + 4), 0, 0);  // global_prefetch_b8
    __builtin_prefetch((const void*)(pb0 + idx + 4), 0, 0);
    c00 = __builtin_amdgcn_wmma_f32_16x16x32_bf16(false, a0.bf, false, b0.bf, (short)0, c00, false, false);
    c01 = __builtin_amdgcn_wmma_f32_16x16x32_bf16(false, a0.bf, false, b1.bf, (short)0, c01, false, false);
    c10 = __builtin_amdgcn_wmma_f32_16x16x32_bf16(false, a1.bf, false, b0.bf, (short)0, c10, false, false);
    c11 = __builtin_amdgcn_wmma_f32_16x16x32_bf16(false, a1.bf, false, b1.bf, (short)0, c11, false, false);
  }
  // C/D layout: vgpr rr -> M = rr + 8*hi, N = lane&15
  const int colLo = n0 + (lane & 15);
  const int colHi = colLo + 16;
  const int rowB  = m0 + (hi ? 8 : 0);
  #pragma unroll
  for (int rr = 0; rr < 8; ++rr) {
    if (colLo < N) {
      C[(size_t)(rowB + rr)      * N + colLo] = c00[rr];
      C[(size_t)(rowB + 16 + rr) * N + colLo] = c10[rr];
    }
    if (colHi < N) {
      C[(size_t)(rowB + rr)      * N + colHi] = c01[rr];
      C[(size_t)(rowB + 16 + rr) * N + colHi] = c11[rr];
    }
  }
}

// --------------------------- elementwise helpers ---------------------------
__global__ void pad_cast_x(const float* __restrict__ x, unsigned short* __restrict__ xp) {
  int idx = blockIdx.x * 256 + threadIdx.x;
  if (idx >= ROWS * KPAD) return;
  int c = idx & (KPAD - 1); int row = idx >> 6;
  float v = (c < INPUT_DIM) ? x[(size_t)row * INPUT_DIM + c] : 0.f;
  xp[idx] = f2bf(v);
}
__global__ void pad_cast_w(const float* __restrict__ w, unsigned short* __restrict__ wp) {
  int idx = blockIdx.x * 256 + threadIdx.x;
  if (idx >= D_MODEL * KPAD) return;
  int c = idx & (KPAD - 1); int rw = idx >> 6;
  float v = (c < INPUT_DIM) ? w[rw * INPUT_DIM + c] : 0.f;
  wp[idx] = f2bf(v);
}
__global__ void cast_bf(const float* __restrict__ s, unsigned short* __restrict__ d, int n) {
  int i = blockIdx.x * 256 + threadIdx.x;
  if (i < n) d[i] = f2bf(s[i]);
}

__global__ __launch_bounds__(256)
void bias_ln(const float* __restrict__ in, const float* __restrict__ bias,
             const float* __restrict__ g, const float* __restrict__ b,
             float* __restrict__ h, unsigned short* __restrict__ hbf) {
  __shared__ float red[256];
  int row = blockIdx.x, c = threadIdx.x;
  float v = in[(size_t)row * D_MODEL + c] + bias[c];
  float mean = block_sum_256(v, red, c) * (1.f / D_MODEL);
  float d = v - mean;
  float var = block_sum_256(d * d, red, c) * (1.f / D_MODEL);
  float o = d * rsqrtf(var + 1e-5f) * g[c] + b[c];
  h  [(size_t)row * D_MODEL + c] = o;
  hbf[(size_t)row * D_MODEL + c] = f2bf(o);
}

__global__ __launch_bounds__(256)
void resid_ln(const float* __restrict__ in, const float* __restrict__ res,
              const float* __restrict__ g, const float* __restrict__ b,
              float* __restrict__ h, unsigned short* __restrict__ hbf) {
  __shared__ float red[256];
  int row = blockIdx.x, c = threadIdx.x;
  float v = in[(size_t)row * D_MODEL + c] + res[(size_t)row * D_MODEL + c];
  float mean = block_sum_256(v, red, c) * (1.f / D_MODEL);
  float d = v - mean;
  float var = block_sum_256(d * d, red, c) * (1.f / D_MODEL);
  float o = d * rsqrtf(var + 1e-5f) * g[c] + b[c];
  h  [(size_t)row * D_MODEL + c] = o;       // in-place over res is safe per-thread
  hbf[(size_t)row * D_MODEL + c] = f2bf(o);
}

// Causal depthwise conv1d (k=4) over xBC slice of zxbcdt, + bias, SiLU.
__global__ void conv_silu(const float* __restrict__ zx, const float* __restrict__ cw,
                          const float* __restrict__ cb, float* __restrict__ xbc) {
  int idx = blockIdx.x * 256 + threadIdx.x;
  if (idx >= ROWS * CONV_DIM) return;
  int c = idx % CONV_DIM;
  int row = idx / CONV_DIM;
  int t = row % SEQLEN;
  float acc = cb[c];
  #pragma unroll
  for (int j = 0; j < D_CONV; ++j) {
    int tt = t - (D_CONV - 1) + j;
    if (tt >= 0)
      acc += zx[(size_t)(row - (D_CONV - 1 - j)) * D_IN_PROJ + D_INNER + c] * cw[c * D_CONV + j];
  }
  xbc[(size_t)row * CONV_DIM + c] = siluf(acc);
}

// Sequential SSM recurrence. One block per (head, batch); state (64x128) f32
// in registers (32/thread: thread t -> p = t>>2, n-slice = (t&3)*32..+32).
__global__ __launch_bounds__(256)
void ssm_scan(const float* __restrict__ zx, const float* __restrict__ xbc,
              float* __restrict__ y,
              const float* __restrict__ dtb, const float* __restrict__ Alog,
              const float* __restrict__ Dp) {
  const int head = blockIdx.x;
  const int b    = blockIdx.y;
  const int tid  = threadIdx.x;
  const int p    = tid >> 2;
  const int nq   = tid & 3;
  const int n0   = nq * 32;
  __shared__ float sB[D_STATE], sC[D_STATE], sx[HEADDIM];
  const float dt_bias = dtb[head];
  const float A  = -__expf(Alog[head]);
  const float Dv = Dp[head];
  float st[32];
  #pragma unroll
  for (int j = 0; j < 32; ++j) st[j] = 0.f;

  for (int t = 0; t < SEQLEN; ++t) {
    const size_t row = (size_t)b * SEQLEN + t;
    const float* xr = xbc + row * CONV_DIM;
    if (tid < D_STATE) sB[tid] = xr[D_INNER + tid];
    else               sC[tid - D_STATE] = xr[D_INNER + D_STATE + (tid - D_STATE)];
    if (tid < HEADDIM) sx[tid] = xr[head * HEADDIM + tid];
    __syncthreads();

    float dtr = zx[row * D_IN_PROJ + (D_INNER + CONV_DIM) + head] + dt_bias;
    float dtv = (dtr > 20.f) ? dtr : log1pf(__expf(dtr));   // softplus
    float dA  = __expf(dtv * A);
    float xv  = sx[p];
    float dtx = dtv * xv;
    float acc = 0.f;
    #pragma unroll
    for (int j = 0; j < 32; ++j) {
      st[j] = st[j] * dA + dtx * sB[n0 + j];
      acc += st[j] * sC[n0 + j];
    }
    acc += __shfl_xor(acc, 1, 32);     // reduce the 4 n-slices (adjacent lanes)
    acc += __shfl_xor(acc, 2, 32);
    if (nq == 0)
      y[row * D_INNER + head * HEADDIM + p] = acc + Dv * xv;
    __syncthreads();
  }
}

// y = y * silu(z); RMS-norm over 512; * norm_w; -> bf16 for out-proj GEMM.
__global__ __launch_bounds__(256)
void gate_rms(const float* __restrict__ y, const float* __restrict__ zx,
              const float* __restrict__ nw, unsigned short* __restrict__ ybf) {
  __shared__ float red[256];
  int row = blockIdx.x, tid = threadIdx.x;
  int c0 = tid, c1 = tid + 256;
  float v0 = y[(size_t)row * D_INNER + c0] * siluf(zx[(size_t)row * D_IN_PROJ + c0]);
  float v1 = y[(size_t)row * D_INNER + c1] * siluf(zx[(size_t)row * D_IN_PROJ + c1]);
  float ms = block_sum_256(v0 * v0 + v1 * v1, red, tid) * (1.f / D_INNER);
  float rr = rsqrtf(ms + 1e-5f);
  ybf[(size_t)row * D_INNER + c0] = f2bf(v0 * rr * nw[c0]);
  ybf[(size_t)row * D_INNER + c1] = f2bf(v1 * rr * nw[c1]);
}

// Classifier head (N=13, tiny) + transpose to (SEQLEN, BATCH, NUM_CLASSES).
__global__ void head_kernel(const float* __restrict__ h, const float* __restrict__ ow,
                            const float* __restrict__ ob, float* __restrict__ out) {
  int idx = blockIdx.x * 256 + threadIdx.x;
  if (idx >= ROWS * NUM_CLASSES) return;
  int c = idx % NUM_CLASSES;
  int row = idx / NUM_CLASSES;
  int b = row / SEQLEN, t = row % SEQLEN;
  float acc = ob[c];
  for (int k = 0; k < D_MODEL; ++k)
    acc += h[(size_t)row * D_MODEL + k] * ow[c * D_MODEL + k];
  out[((size_t)t * BATCH + b) * NUM_CLASSES + c] = acc;
}

// ---------------------------------------------------------------------------
extern "C" void kernel_launch(void* const* d_in, const int* in_sizes, int n_in,
                              void* d_out, int out_size, void* d_ws, size_t ws_size,
                              hipStream_t stream) {
  const float* x       = (const float*)d_in[0];
  const float* in_w    = (const float*)d_in[1];
  const float* in_b    = (const float*)d_in[2];
  const float* ln_in_g = (const float*)d_in[3];
  const float* ln_in_b = (const float*)d_in[4];
  const float* inproj  = (const float*)d_in[5];
  const float* convw   = (const float*)d_in[6];
  const float* convb   = (const float*)d_in[7];
  const float* dtb     = (const float*)d_in[8];
  const float* Alog    = (const float*)d_in[9];
  const float* Dp      = (const float*)d_in[10];
  const float* normw   = (const float*)d_in[11];
  const float* outproj = (const float*)d_in[12];
  const float* ln_g    = (const float*)d_in[13];
  const float* ln_b    = (const float*)d_in[14];
  const float* out_w   = (const float*)d_in[15];
  const float* out_b   = (const float*)d_in[16];
  float* out = (float*)d_out;

  char* ws = (char*)d_ws;
  size_t off = 0;
  auto alloc = [&](size_t bytes) -> char* {
    char* p = ws + off;
    off += (bytes + 255) & ~(size_t)255;
    return p;
  };
  float*          hF  = (float*)         alloc((size_t)ROWS * D_MODEL * 4);
  unsigned short* hB  = (unsigned short*)alloc((size_t)ROWS * D_MODEL * 2);
  float*          tmp = (float*)         alloc((size_t)ROWS * D_MODEL * 4);
  float*          zx  = (float*)         alloc((size_t)ROWS * D_IN_PROJ * 4);
  float*          xbc = (float*)         alloc((size_t)ROWS * CONV_DIM * 4);
  float*          yF  = (float*)         alloc((size_t)ROWS * D_INNER * 4);
  unsigned short* yB  = (unsigned short*)alloc((size_t)ROWS * D_INNER * 2);
  unsigned short* xp  = (unsigned short*)alloc((size_t)ROWS * KPAD * 2);
  unsigned short* wp  = (unsigned short*)alloc((size_t)D_MODEL * KPAD * 2);
  unsigned short* wbf = (unsigned short*)alloc((size_t)D_IN_PROJ * D_MODEL * 2);

  // Embed: h = layernorm(x @ in_w^T + in_b)   (K padded 63 -> 64)
  pad_cast_x<<<(ROWS * KPAD + 255) / 256, 256, 0, stream>>>(x, xp);
  pad_cast_w<<<(D_MODEL * KPAD + 255) / 256, 256, 0, stream>>>(in_w, wp);
  gemm_bf16_wmma<<<dim3((D_MODEL + 63) / 64, ROWS / 128), 256, 0, stream>>>(xp, wp, tmp, D_MODEL, KPAD);
  bias_ln<<<ROWS, 256, 0, stream>>>(tmp, in_b, ln_in_g, ln_in_b, hF, hB);

  for (int l = 0; l < NUM_LAYERS; ++l) {
    cast_bf<<<(D_IN_PROJ * D_MODEL + 255) / 256, 256, 0, stream>>>(
        inproj + (size_t)l * D_IN_PROJ * D_MODEL, wbf, D_IN_PROJ * D_MODEL);
    gemm_bf16_wmma<<<dim3((D_IN_PROJ + 63) / 64, ROWS / 128), 256, 0, stream>>>(
        hB, wbf, zx, D_IN_PROJ, D_MODEL);
    conv_silu<<<(ROWS * CONV_DIM + 255) / 256, 256, 0, stream>>>(
        zx, convw + (size_t)l * CONV_DIM * D_CONV, convb + (size_t)l * CONV_DIM, xbc);
    ssm_scan<<<dim3(NHEADS, BATCH), 256, 0, stream>>>(
        zx, xbc, yF, dtb + l * NHEADS, Alog + l * NHEADS, Dp + l * NHEADS);
    gate_rms<<<ROWS, 256, 0, stream>>>(yF, zx, normw + (size_t)l * D_INNER, yB);
    cast_bf<<<(D_MODEL * D_INNER + 255) / 256, 256, 0, stream>>>(
        outproj + (size_t)l * D_MODEL * D_INNER, wbf, D_MODEL * D_INNER);
    gemm_bf16_wmma<<<dim3((D_MODEL + 63) / 64, ROWS / 128), 256, 0, stream>>>(
        yB, wbf, tmp, D_MODEL, D_INNER);
    resid_ln<<<ROWS, 256, 0, stream>>>(
        tmp, hF, ln_g + (size_t)l * D_MODEL, ln_b + (size_t)l * D_MODEL, hF, hB);
  }

  head_kernel<<<(ROWS * NUM_CLASSES + 255) / 256, 256, 0, stream>>>(hF, out_w, out_b, out);
}
